// HybridReasoningModel_5712306504127
// MI455X (gfx1250) — compile-verified, matrix operand order checked by reading, not run on hardware
//
#include <hip/hip_runtime.h>
#include <hip/hip_bf16.h>
#include <math.h>

// ---------------------------------------------------------------- constants
#define DM    256
#define NH    4
#define HDIM  64
#define TSEQ  512
#define BSZ   4
#define NLAY  3
#define MROWS (BSZ*TSEQ)          // 2048
#define VOCAB 50257
#define VPAD  50304               // VOCAB rounded up to 64

typedef __attribute__((ext_vector_type(16))) __bf16 v16bf;
typedef __attribute__((ext_vector_type(8)))  __bf16 v8bf;
typedef __attribute__((ext_vector_type(8)))  float  v8f;

// CDNA5 16-bit A/B fragment: lane L holds row/col (L&15); half-wave (L>>4)
// takes k-chunks [h*8, h*8+8) and [16+h*8, 16+h*8+8) -> two 16B loads.
__device__ inline v16bf frag16(const __bf16* p) {
  union { v16bf v; v8bf h[2]; } u;
  u.h[0] = *(const v8bf*)(p);
  u.h[1] = *(const v8bf*)(p + 16);
  return u.v;
}
__device__ inline v8f zero8() { v8f z = {0.f,0.f,0.f,0.f,0.f,0.f,0.f,0.f}; return z; }
__device__ inline float gelu_f(float x) { return 0.5f*x*(1.f + erff(x*0.70710678118f)); }

// ---------------------------------------------------------------- TDM
#if __has_builtin(__builtin_amdgcn_tensor_load_to_lds)
#define HAVE_TDM 1
#else
#define HAVE_TDM 0
#endif

typedef __attribute__((ext_vector_type(4))) unsigned int tdm_u4;
typedef __attribute__((ext_vector_type(8))) int          tdm_i8;
typedef __attribute__((ext_vector_type(4))) int          tdm_i4;

#if HAVE_TDM
// 2D tile load: tileN rows x tileK contiguous bf16 elems, row stride strideK
// elems, into LDS at lds_byte_off. D# layout per CDNA5 ISA ch.8 (group0/1).
__device__ inline void tdm_load_2d(unsigned lds_byte_off, const __bf16* gsrc,
                                   unsigned tileK, unsigned tileN,
                                   unsigned long long tensorK,
                                   unsigned long long tensorN,
                                   unsigned long long strideK) {
  unsigned long long ga = (unsigned long long)(uintptr_t)gsrc;
  tdm_u4 g0;
  g0[0] = 1u;                                          // count=1, user D#
  g0[1] = lds_byte_off;                                // lds_addr
  g0[2] = (unsigned)(ga & 0xffffffffu);                // global_addr[31:0]
  g0[3] = (unsigned)((ga >> 32) & 0x01ffffffu)         // global_addr[56:32]
        | 0x80000000u;                                 // type=2 ("image")
  tdm_i8 g1;
  g1[0] = (int)(1u << 16);                             // data_size=1 (2 bytes)
  g1[1] = (int)((tensorK & 0xffffu) << 16);            // tensor_dim0[15:0]
  g1[2] = (int)(((tensorK >> 16) & 0xffffu) | ((tensorN & 0xffffu) << 16));
  g1[3] = (int)(((tensorN >> 16) & 0xffffu) | ((tileK & 0xffffu) << 16));
  g1[4] = (int)(tileN & 0xffffu);                      // tile_dim1 (tile_dim2=0)
  g1[5] = (int)(strideK & 0xffffffffu);                // tensor_dim0_stride lo
  g1[6] = (int)((strideK >> 32) & 0xffffu);            // stride hi (dim1_stride=0)
  g1[7] = 0;
  tdm_i4 z4 = {0,0,0,0};
#if __clang_major__ >= 23
  tdm_i8 z8 = {0,0,0,0,0,0,0,0};
  __builtin_amdgcn_tensor_load_to_lds(g0, g1, z4, z4, z8, 0);
#else
  __builtin_amdgcn_tensor_load_to_lds(g0, g1, z4, z4, 0);
#endif
}
#endif

// ------------------------------------------------- weight transpose->bf16
// W fp32 [K][N] -> Wt bf16 [Npad][K] (rows >= N replicate row N-1)
__global__ void k_trans(const float* __restrict__ W, __bf16* __restrict__ Wt,
                        int K, int N, int Npad) {
  long i = (long)blockIdx.x*256 + threadIdx.x;
  if (i >= (long)K*Npad) return;
  int n = (int)(i % Npad), k = (int)(i / Npad);
  int ns = n < N ? n : N - 1;
  Wt[(size_t)n*K + k] = (__bf16)W[(size_t)k*N + ns];
}

// ------------------------------------------------- embedding + pos encode
__global__ void k_embed(const int* __restrict__ x, const float* __restrict__ emb,
                        float* __restrict__ prf, __bf16* __restrict__ prb) {
  int i = blockIdx.x*256 + threadIdx.x;          // B*T*D
  int d = i & 255; int bt = i >> 8; int t = bt & 511;
  int tok = x[bt];
  float divv = __expf(-9.210340372f/256.f * (float)(d & ~1));
  float ang  = (float)t * divv;
  float pe   = (d & 1) ? __cosf(ang) : __sinf(ang);
  float v = emb[(size_t)tok*DM + d] + pe;
  prf[i] = v; prb[i] = (__bf16)v;
}

// ------------------------------------------------- register-direct WMMA GEMM
// (small-M path; used for the M=4 s2p projection)
template <int ACT, typename TO>
__global__ __launch_bounds__(32)
void k_gemm(const __bf16* __restrict__ A, const __bf16* __restrict__ Wt,
            const float* __restrict__ bias, TO* __restrict__ C,
            int M, int N, int K) {
  const int lane = threadIdx.x;
  const int hlf  = lane >> 4, l16 = lane & 15;
  const int row0 = blockIdx.y * 16;
  const int col0 = blockIdx.x * 64;
  int ar = row0 + l16; if (ar >= M) ar = M - 1;
  const __bf16* Arow = A + (size_t)ar*K;
  v8f acc[4] = { zero8(), zero8(), zero8(), zero8() };
  for (int kc = 0; kc < K; kc += 32) {
    v16bf a = frag16(Arow + kc + hlf*8);
#pragma unroll
    for (int j = 0; j < 4; ++j) {
      int cn = col0 + j*16 + l16; if (cn >= N) cn = N - 1;
      v16bf b = frag16(Wt + (size_t)cn*K + kc + hlf*8);
      acc[j] = __builtin_amdgcn_wmma_f32_16x16x32_bf16(
          false, a, false, b, (short)0, acc[j], false, false);
    }
  }
#pragma unroll
  for (int j = 0; j < 4; ++j) {
    int cn = col0 + j*16 + l16;
    if (cn >= N) continue;
    float bv = bias ? bias[cn] : 0.f;
#pragma unroll
    for (int v = 0; v < 8; ++v) {
      int r = row0 + v + hlf*8;                  // C layout: M = v + 8*half
      if (r >= M) continue;
      float xv = acc[j][v] + bv;
      if (ACT == 1) xv = gelu_f(xv);
      C[(size_t)r*N + cn] = (TO)xv;
    }
  }
}

// ------------------------------------------------- TDM-staged WMMA GEMM
// 4 waves, 64x64 C tile. B tile (64 cols x 64 k) staged once per block into
// LDS via TENSOR_LOAD_TO_LDS and shared by all waves (4x less L2 traffic).
// Nw = padded row count of Wt (multiple of 64); stores bound-check against N.
template <int ACT, typename TO>
__global__ __launch_bounds__(128)
void k_gemm_tdm(const __bf16* __restrict__ A, const __bf16* __restrict__ Wt,
                const float* __restrict__ bias, TO* __restrict__ C,
                int M, int N, int K, int Nw) {
  __shared__ __align__(16) __bf16 Bt[64*64];
  const int wave = threadIdx.x >> 5;
  const int lane = threadIdx.x & 31;
  const int hlf  = lane >> 4, l16 = lane & 15;
  const int row0 = blockIdx.y*64 + wave*16;
  const int col0 = blockIdx.x*64;
  int ar = row0 + l16; if (ar >= M) ar = M - 1;
  const __bf16* Arow = A + (size_t)ar*K;
  v8f acc[4] = { zero8(), zero8(), zero8(), zero8() };
  for (int kc = 0; kc < K; kc += 64) {
#if HAVE_TDM
    if (wave == 0) {
      tdm_load_2d((unsigned)(unsigned long long)(uintptr_t)(void*)Bt,
                  Wt + (size_t)col0*K + kc,
                  64u, 64u,
                  (unsigned long long)K,
                  (unsigned long long)(Nw - col0),
                  (unsigned long long)K);
      __builtin_amdgcn_s_wait_tensorcnt(0);
    }
#else
    for (int i = threadIdx.x; i < 64*64/8; i += 128) {
      int rrow = i >> 3, cc = i & 7;
      *(v8bf*)(Bt + rrow*64 + cc*8) =
          *(const v8bf*)(Wt + (size_t)(col0 + rrow)*K + kc + cc*8);
    }
#endif
    __syncthreads();
#pragma unroll
    for (int kk = 0; kk < 64; kk += 32) {
      v16bf a = frag16(Arow + kc + kk + hlf*8);
#pragma unroll
      for (int j = 0; j < 4; ++j) {
        v16bf b = frag16(Bt + (j*16 + l16)*64 + kk + hlf*8);
        acc[j] = __builtin_amdgcn_wmma_f32_16x16x32_bf16(
            false, a, false, b, (short)0, acc[j], false, false);
      }
    }
    __syncthreads();
  }
#pragma unroll
  for (int j = 0; j < 4; ++j) {
    int cn = col0 + j*16 + l16;
    if (cn >= N) continue;
    float bv = bias ? bias[cn] : 0.f;
#pragma unroll
    for (int v = 0; v < 8; ++v) {
      int r = row0 + v + hlf*8;
      if (r >= M) continue;
      float xv = acc[j][v] + bv;
      if (ACT == 1) xv = gelu_f(xv);
      C[(size_t)r*N + cn] = (TO)xv;
    }
  }
}

// ------------------------------------------------- head packing
__global__ void k_pack_qk(const float* __restrict__ s, __bf16* __restrict__ d) {
  int i = blockIdx.x*256 + threadIdx.x;
  int hd = i & 63, h = (i>>6)&3, t = (i>>8)&511, b = i>>17;
  d[(((size_t)(b*NH+h))*TSEQ + t)*HDIM + hd] = (__bf16)s[i];
}
__global__ void k_pack_vt(const float* __restrict__ s, __bf16* __restrict__ d) {
  int i = blockIdx.x*256 + threadIdx.x;
  int hd = i & 63, h = (i>>6)&3, t = (i>>8)&511, b = i>>17;
  d[(((size_t)(b*NH+h))*HDIM + hd)*TSEQ + t] = (__bf16)s[i];
}

// ------------------------------------------------- causal attention (1 wave)
__global__ __launch_bounds__(32)
void k_attn(const __bf16* __restrict__ Qb, const __bf16* __restrict__ Kb,
            const __bf16* __restrict__ Vt, __bf16* __restrict__ ctx) {
  __shared__ __align__(16) float  sc[16][TSEQ];
  __shared__ __align__(16) __bf16 pb[16][TSEQ];
  const int lane = threadIdx.x, hlf = lane >> 4, l16 = lane & 15;
  const int qt = blockIdx.x & 31;
  const int bh = blockIdx.x >> 5;
  const int b  = bh >> 2, h = bh & 3;
  const int q0 = qt * 16;
  const __bf16* Qp = Qb + ((size_t)bh*TSEQ + q0)*HDIM;
  v16bf aq0 = frag16(Qp + (size_t)l16*HDIM + 0  + hlf*8);
  v16bf aq1 = frag16(Qp + (size_t)l16*HDIM + 32 + hlf*8);
  const int ntile = qt + 1;
  for (int st = 0; st < ntile; ++st) {
    int s0 = st * 16;
    v8f acc = zero8();
    v16bf bk0 = frag16(Kb + ((size_t)bh*TSEQ + s0 + l16)*HDIM + 0  + hlf*8);
    acc = __builtin_amdgcn_wmma_f32_16x16x32_bf16(false, aq0, false, bk0, (short)0, acc, false, false);
    v16bf bk1 = frag16(Kb + ((size_t)bh*TSEQ + s0 + l16)*HDIM + 32 + hlf*8);
    acc = __builtin_amdgcn_wmma_f32_16x16x32_bf16(false, aq1, false, bk1, (short)0, acc, false, false);
    int s = s0 + l16;
#pragma unroll
    for (int v = 0; v < 8; ++v) {
      int q = q0 + v + hlf*8;
      float val = acc[v] * 0.125f;
      if (s > q) val = -1e9f;
      sc[v + hlf*8][s] = val;
    }
  }
  __syncthreads();
  const int len  = ntile * 16;
  const int lenp = (len + 31) & ~31;
  if (lane < 16) {
    float m = -1e30f;
    for (int s = 0; s < len; ++s) m = fmaxf(m, sc[lane][s]);
    float sum = 0.f;
    for (int s = 0; s < len; ++s) { float e = __expf(sc[lane][s]-m); sc[lane][s]=e; sum+=e; }
    float inv = 1.f / sum;
    for (int s = 0; s < len; ++s)   pb[lane][s] = (__bf16)(sc[lane][s]*inv);
    for (int s = len; s < lenp; ++s) pb[lane][s] = (__bf16)0.f;
  }
  __syncthreads();
  v8f cacc[4] = { zero8(), zero8(), zero8(), zero8() };
  for (int kc = 0; kc < lenp; kc += 32) {
    v16bf ap = frag16(&pb[l16][kc + hlf*8]);
#pragma unroll
    for (int j = 0; j < 4; ++j) {
      v16bf bv = frag16(Vt + ((size_t)bh*HDIM + j*16 + l16)*TSEQ + kc + hlf*8);
      cacc[j] = __builtin_amdgcn_wmma_f32_16x16x32_bf16(false, ap, false, bv, (short)0, cacc[j], false, false);
    }
  }
#pragma unroll
  for (int j = 0; j < 4; ++j)
#pragma unroll
    for (int v = 0; v < 8; ++v) {
      int q  = q0 + v + hlf*8;
      int hd = j*16 + l16;
      ctx[((size_t)b*TSEQ + q)*DM + h*HDIM + hd] = (__bf16)cacc[j][v];
    }
}

// ------------------------------------------------- residual add + layernorm
__global__ void k_addln(const float* __restrict__ x, const float* __restrict__ y,
                        const float* __restrict__ g, const float* __restrict__ bta,
                        float* __restrict__ outf, __bf16* __restrict__ outb) {
  const int row = blockIdx.x, tid = threadIdx.x;
  __shared__ float sm[256];
  float v = x[(size_t)row*DM + tid];
  if (y) v += y[(size_t)row*DM + tid];
  sm[tid] = v; __syncthreads();
  for (int off = 128; off > 0; off >>= 1) { if (tid < off) sm[tid] += sm[tid+off]; __syncthreads(); }
  float mu = sm[0] * (1.f/DM); __syncthreads();
  float d = v - mu;
  sm[tid] = d*d; __syncthreads();
  for (int off = 128; off > 0; off >>= 1) { if (tid < off) sm[tid] += sm[tid+off]; __syncthreads(); }
  float var = sm[0] * (1.f/DM);
  float o = d * rsqrtf(var + 1e-5f) * g[tid] + bta[tid];
  if (outf) outf[(size_t)row*DM + tid] = o;
  if (outb) outb[(size_t)row*DM + tid] = (__bf16)o;
}

// ------------------------------------------------- ffn input: pr + 0.3*sg[b]
__global__ void k_ffnin(const float* __restrict__ pr, const float* __restrict__ sg,
                        __bf16* __restrict__ out) {
  int i = blockIdx.x*256 + threadIdx.x;
  int d = i & 255, bt = i >> 8, b = bt >> 9;
  out[i] = (__bf16)(pr[i] + 0.3f*sg[b*DM + d]);
}

__global__ void k_init_h(float* hf, __bf16* hb) {
  int i = blockIdx.x*256 + threadIdx.x;
  if (i < BSZ*DM) { hf[i] = 0.f; hb[i] = (__bf16)0.f; }
}
__global__ void k_copy_h(const float* __restrict__ hf, float* __restrict__ out) {
  int i = blockIdx.x*256 + threadIdx.x;
  if (i < BSZ*DM) out[i] = hf[i];
}

// ------------------------------------------------- scan helper: padded-M WMMA
template <typename TO>
__device__ inline void gemm16_rows4(const __bf16* Ab, const __bf16* __restrict__ Wt,
                                    const float* __restrict__ bias, TO* out,
                                    int N, int K, int act) {
  const int wave = threadIdx.x >> 5;
  const int lane = threadIdx.x & 31;
  const int hlf  = lane >> 4, l16 = lane & 15;
  const int r4   = l16 & 3;
  for (int n0 = wave*16; n0 < N; n0 += 128) {
    v8f acc = zero8();
    for (int kc = 0; kc < K; kc += 32) {
      v16bf a = frag16(Ab + r4*K + kc + hlf*8);
      v16bf b = frag16(Wt + (size_t)(n0 + l16)*K + kc + hlf*8);
      acc = __builtin_amdgcn_wmma_f32_16x16x32_bf16(false, a, false, b, (short)0, acc, false, false);
    }
    if (hlf == 0) {
      float bv = bias ? bias[n0 + l16] : 0.f;
#pragma unroll
      for (int v = 0; v < 4; ++v) {
        float xv = acc[v] + bv;
        if (act) xv = gelu_f(xv);
        out[v*N + n0 + l16] = (TO)xv;
      }
    }
  }
}

// ------------------------------------------------- persistent sequential scan
__global__ __launch_bounds__(256)
void k_scan(const float* __restrict__ pr,
            const __bf16* taWq, const __bf16* taWk, const __bf16* taWv,
            const __bf16* taWo, const __bf16* p2sW,
            const __bf16* trW1, const __bf16* trW2,
            const float* ta_bq, const float* ta_bk, const float* ta_bv,
            const float* ta_bo, const float* p2sb,
            const float* tr_b1, const float* tr_b2,
            const float* ln3g, const float* ln3b,
            const float* ln4g, const float* ln4b,
            float* __restrict__ hf_g, __bf16* __restrict__ hb_g,
            float* __restrict__ kbuf, float* __restrict__ vbuf) {
  __shared__ __align__(16) float  h[BSZ*DM];
  __shared__ __align__(16) float  xf[BSZ*DM];
  __shared__ __align__(16) __bf16 xb[BSZ*DM];
  __shared__ __align__(16) float  qd[BSZ*DM];
  __shared__ __align__(16) float  t0[BSZ*DM];
  __shared__ __align__(16) __bf16 t0b[BSZ*DM];
  __shared__ __align__(16) float  t1[BSZ*DM];
  __shared__ __align__(16) float  sc[16*TSEQ];
  __shared__ float red2[256];
  __shared__ float smax[16];
  const int tid = threadIdx.x;
  for (int i = tid; i < BSZ*DM; i += 256) h[i] = hf_g[i];
  __syncthreads();

  for (int t = 0; t < TSEQ; ++t) {
    for (int i = tid; i < BSZ*DM; i += 256) {
      int b = i >> 8, d = i & 255;
      float v = pr[((size_t)b*TSEQ + t)*DM + d];
      xf[i] = v; xb[i] = (__bf16)v;
    }
    __syncthreads();
    gemm16_rows4<float>(xb, taWq, ta_bq, qd, DM, DM, 0);
    __syncthreads();

    if (t > 0) {
      const int p = tid >> 4, l = tid & 15;
      const int b = p >> 2, hh = p & 3;
      const float* qr = qd + b*DM + hh*HDIM;
      float m = -1e30f;
      for (int s = l; s < t; s += 16) {
        const float* kr = kbuf + (size_t)s*(BSZ*DM) + b*DM + hh*HDIM;
        float dsum = 0.f;
#pragma unroll 8
        for (int e = 0; e < HDIM; ++e) dsum += qr[e]*kr[e];
        dsum *= 0.125f;
        sc[p*TSEQ + s] = dsum;
        m = fmaxf(m, dsum);
      }
      red2[tid] = m; __syncthreads();
      if (l == 0) { float mm=-1e30f; for (int i2=0;i2<16;++i2) mm=fmaxf(mm,red2[p*16+i2]); smax[p]=mm; }
      __syncthreads();
      float mm = smax[p], ss = 0.f;
      for (int s = l; s < t; s += 16) { float e=__expf(sc[p*TSEQ+s]-mm); sc[p*TSEQ+s]=e; ss+=e; }
      red2[tid] = ss; __syncthreads();
      if (l == 0) { float tt=0.f; for (int i2=0;i2<16;++i2) tt+=red2[p*16+i2]; smax[p]=1.f/tt; }
      __syncthreads();
      float inv = smax[p];
      float a0=0.f,a1=0.f,a2=0.f,a3=0.f;
      for (int s = 0; s < t; ++s) {
        float w = sc[p*TSEQ+s]*inv;
        const float* vr = vbuf + (size_t)s*(BSZ*DM) + b*DM + hh*HDIM + l*4;
        a0 += w*vr[0]; a1 += w*vr[1]; a2 += w*vr[2]; a3 += w*vr[3];
      }
      int o = b*DM + hh*HDIM + l*4;
      t0[o+0]=a0; t0[o+1]=a1; t0[o+2]=a2; t0[o+3]=a3;
      __syncthreads();
      for (int i = tid; i < BSZ*DM; i += 256) t0b[i] = (__bf16)t0[i];
      __syncthreads();
      gemm16_rows4<float>(t0b, taWo, ta_bo, t1, DM, DM, 0);
    } else {
      for (int i = tid; i < BSZ*DM; i += 256) t1[i] = 0.f;
    }
    __syncthreads();
    gemm16_rows4<float>(xb, p2sW, p2sb, t0, DM, DM, 0);
    __syncthreads();
    __bf16* combb = (__bf16*)sc;
    for (int i = tid; i < BSZ*DM; i += 256) {
      int b = i >> 8, d = i & 255;
      combb[b*2*DM + d]      = (__bf16)(xf[i] + 0.3f*t0[i]);
      combb[b*2*DM + DM + d] = (__bf16)t1[i];
    }
    __syncthreads();
    __bf16* midb = (__bf16*)(sc + 1024);
    gemm16_rows4<__bf16>(combb, trW1, tr_b1, midb, 2*DM, 2*DM, 1);
    __syncthreads();
    gemm16_rows4<float>(midb, trW2, tr_b2, qd, DM, 2*DM, 0);
    __syncthreads();
    {
      const int rr = tid >> 6, c = tid & 63;
      float vals[4]; float ps = 0.f;
#pragma unroll
      for (int e = 0; e < 4; ++e) { float nv = h[rr*DM+c*4+e] + 0.5f*qd[rr*DM+c*4+e]; vals[e]=nv; ps+=nv; }
      red2[tid] = ps; __syncthreads();
      if (c == 0) { float s2=0.f; for (int i2=0;i2<64;++i2) s2+=red2[rr*64+i2]; smax[rr]=s2*(1.f/DM); }
      __syncthreads();
      float mu = smax[rr]; float pv = 0.f;
#pragma unroll
      for (int e = 0; e < 4; ++e) { float dd = vals[e]-mu; pv += dd*dd; }
      red2[tid] = pv; __syncthreads();
      if (c == 0) { float s2=0.f; for (int i2=0;i2<64;++i2) s2+=red2[rr*64+i2]; smax[8+rr]=s2*(1.f/DM); }
      __syncthreads();
      float inv = rsqrtf(smax[8+rr] + 1e-5f);
#pragma unroll
      for (int e = 0; e < 4; ++e) { int d=c*4+e; h[rr*DM+d] = (vals[e]-mu)*inv*ln3g[d] + ln3b[d]; }
    }
    __syncthreads();
    for (int i = tid; i < BSZ*DM; i += 256) t0b[i] = (__bf16)h[i];
    __syncthreads();
    gemm16_rows4<float>(t0b, taWk, ta_bk, t0, DM, DM, 0);
    gemm16_rows4<float>(t0b, taWv, ta_bv, t1, DM, DM, 0);
    __syncthreads();
    for (int i = tid; i < BSZ*DM; i += 256) {
      kbuf[(size_t)t*(BSZ*DM) + i] = t0[i];
      vbuf[(size_t)t*(BSZ*DM) + i] = t1[i];
    }
    __syncthreads();
  }
  {
    const int rr = tid >> 6, c = tid & 63;
    float vals[4]; float ps = 0.f;
#pragma unroll
    for (int e = 0; e < 4; ++e) { float nv = h[rr*DM+c*4+e]; vals[e]=nv; ps+=nv; }
    red2[tid] = ps; __syncthreads();
    if (c == 0) { float s2=0.f; for (int i2=0;i2<64;++i2) s2+=red2[rr*64+i2]; smax[rr]=s2*(1.f/DM); }
    __syncthreads();
    float mu = smax[rr]; float pv = 0.f;
#pragma unroll
    for (int e = 0; e < 4; ++e) { float dd = vals[e]-mu; pv += dd*dd; }
    red2[tid] = pv; __syncthreads();
    if (c == 0) { float s2=0.f; for (int i2=0;i2<64;++i2) s2+=red2[rr*64+i2]; smax[8+rr]=s2*(1.f/DM); }
    __syncthreads();
    float inv = rsqrtf(smax[8+rr] + 1e-5f);
#pragma unroll
    for (int e = 0; e < 4; ++e) {
      int d = c*4+e;
      float o = (vals[e]-mu)*inv*ln4g[d] + ln4b[d];
      hf_g[rr*DM+d] = o; hb_g[rr*DM+d] = (__bf16)o;
    }
  }
}

// ============================================================== launcher
extern "C" void kernel_launch(void* const* d_in, const int* in_sizes, int n_in,
                              void* d_out, int out_size, void* d_ws, size_t ws_size,
                              hipStream_t stream) {
  (void)in_sizes; (void)n_in; (void)out_size; (void)ws_size;
  const float* emb = (const float*)d_in[0];
  auto LW = [&](int l, int j) { return (const float*)d_in[1 + l*36 + j]; };
  const int base = 1 + NLAY*36;
  const float* lnf_g = (const float*)d_in[base + 0];
  const float* lnf_b = (const float*)d_in[base + 1];
  const float* dec_W = (const float*)d_in[base + 2];
  const float* dec_b = (const float*)d_in[base + 3];
  const int*   x     = (const int*)  d_in[base + 4];

  char* wsp = (char*)d_ws;
  auto alloc = [&](size_t bytes) { void* p = wsp; wsp += (bytes + 255) & ~(size_t)255; return p; };

  __bf16 *saWqT[NLAY], *saWkT[NLAY], *saWvT[NLAY], *saWoT[NLAY];
  __bf16 *f1T[NLAY], *f2T[NLAY];
  __bf16 *taWqT[NLAY], *taWkT[NLAY], *taWvT[NLAY], *taWoT[NLAY];
  __bf16 *tr1T[NLAY], *tr2T[NLAY], *p2sT[NLAY], *s2pT[NLAY];
  for (int l = 0; l < NLAY; ++l) {
    saWqT[l]=(__bf16*)alloc(DM*DM*2);   saWkT[l]=(__bf16*)alloc(DM*DM*2);
    saWvT[l]=(__bf16*)alloc(DM*DM*2);   saWoT[l]=(__bf16*)alloc(DM*DM*2);
    f1T[l]  =(__bf16*)alloc(DM*4*DM*2); f2T[l]  =(__bf16*)alloc(4*DM*DM*2);
    taWqT[l]=(__bf16*)alloc(DM*DM*2);   taWkT[l]=(__bf16*)alloc(DM*DM*2);
    taWvT[l]=(__bf16*)alloc(DM*DM*2);   taWoT[l]=(__bf16*)alloc(DM*DM*2);
    tr1T[l] =(__bf16*)alloc(2*DM*2*DM*2); tr2T[l]=(__bf16*)alloc(2*DM*DM*2);
    p2sT[l] =(__bf16*)alloc(DM*DM*2);   s2pT[l] =(__bf16*)alloc(DM*DM*2);
  }
  __bf16* decWT = (__bf16*)alloc((size_t)VPAD*DM*2);

  float*  prf    = (float*) alloc((size_t)MROWS*DM*4);
  __bf16* prb    = (__bf16*)alloc((size_t)MROWS*DM*2);
  float*  qf     = (float*) alloc((size_t)MROWS*DM*4);
  float*  kf     = (float*) alloc((size_t)MROWS*DM*4);
  float*  vf     = (float*) alloc((size_t)MROWS*DM*4);
  __bf16* Qb     = (__bf16*)alloc((size_t)MROWS*DM*2);
  __bf16* Kb     = (__bf16*)alloc((size_t)MROWS*DM*2);
  __bf16* Vtb    = (__bf16*)alloc((size_t)MROWS*DM*2);
  __bf16* ctxb   = (__bf16*)alloc((size_t)MROWS*DM*2);
  float*  attnf  = (float*) alloc((size_t)MROWS*DM*4);
  float*  sgf    = (float*) alloc(BSZ*DM*4);
  __bf16* ffninb = (__bf16*)alloc((size_t)MROWS*DM*2);
  __bf16* midb   = (__bf16*)alloc((size_t)MROWS*4*DM*2);
  float*  ffnf   = (float*) alloc((size_t)MROWS*DM*4);
  float*  lnff   = (float*) alloc((size_t)MROWS*DM*4);
  __bf16* lnfb   = (__bf16*)alloc((size_t)MROWS*DM*2);
  float*  hf_g   = (float*) alloc(BSZ*DM*4);
  __bf16* hb_g   = (__bf16*)alloc(BSZ*DM*2);
  float*  kbufg  = (float*) alloc((size_t)TSEQ*BSZ*DM*4);
  float*  vbufg  = (float*) alloc((size_t)TSEQ*BSZ*DM*4);

  auto trans = [&](const float* W, __bf16* Wt, int K, int N, int Npad) {
    long n = (long)K*Npad;
    k_trans<<<dim3((unsigned)((n+255)/256)), 256, 0, stream>>>(W, Wt, K, N, Npad);
  };
  for (int l = 0; l < NLAY; ++l) {
    trans(LW(l,0),  saWqT[l], DM, DM, DM);   trans(LW(l,1),  saWkT[l], DM, DM, DM);
    trans(LW(l,2),  saWvT[l], DM, DM, DM);   trans(LW(l,6),  saWoT[l], DM, DM, DM);
    trans(LW(l,8),  f1T[l],   DM, 4*DM, 4*DM); trans(LW(l,10), f2T[l], 4*DM, DM, DM);
    trans(LW(l,12), taWqT[l], DM, DM, DM);   trans(LW(l,13), taWkT[l], DM, DM, DM);
    trans(LW(l,14), taWvT[l], DM, DM, DM);   trans(LW(l,18), taWoT[l], DM, DM, DM);
    trans(LW(l,20), tr1T[l],  2*DM, 2*DM, 2*DM); trans(LW(l,22), tr2T[l], 2*DM, DM, DM);
    trans(LW(l,24), p2sT[l],  DM, DM, DM);   trans(LW(l,26), s2pT[l],  DM, DM, DM);
  }
  trans(dec_W, decWT, DM, VOCAB, VPAD);

  // big-M GEMMs: TDM-staged B tiles
  auto gemm_big = [&](const __bf16* A, const __bf16* Wt, const float* bias,
                      float* C, int M, int N, int K, int Nw) {
    dim3 g(Nw/64, (M+63)/64);
    k_gemm_tdm<0,float><<<g, 128, 0, stream>>>(A, Wt, bias, C, M, N, K, Nw);
  };

  k_embed<<<MROWS, 256, 0, stream>>>(x, emb, prf, prb);
  k_init_h<<<4, 256, 0, stream>>>(hf_g, hb_g);

  for (int l = 0; l < NLAY; ++l) {
    // ---- causal self-attention
    gemm_big(prb, saWqT[l], LW(l,3), qf, MROWS, DM, DM, DM);
    gemm_big(prb, saWkT[l], LW(l,4), kf, MROWS, DM, DM, DM);
    gemm_big(prb, saWvT[l], LW(l,5), vf, MROWS, DM, DM, DM);
    k_pack_qk<<<MROWS, 256, 0, stream>>>(qf, Qb);
    k_pack_qk<<<MROWS, 256, 0, stream>>>(kf, Kb);
    k_pack_vt<<<MROWS, 256, 0, stream>>>(vf, Vtb);
    k_attn<<<BSZ*NH*(TSEQ/16), 32, 0, stream>>>(Qb, Kb, Vtb, ctxb);
    gemm_big(ctxb, saWoT[l], LW(l,7), attnf, MROWS, DM, DM, DM);
    k_addln<<<MROWS, 256, 0, stream>>>(prf, attnf, LW(l,28), LW(l,29), prf, prb);
    // ---- FFN with state guidance
    {
      dim3 g(DM/64, 1);   // M=4 s2p projection: small-M register-direct path
      k_gemm<0,float><<<g, 32, 0, stream>>>(hb_g, s2pT[l], LW(l,27), sgf, BSZ, DM, DM);
    }
    k_ffnin<<<MROWS, 256, 0, stream>>>(prf, sgf, ffninb);
    {
      dim3 g((4*DM)/64, MROWS/64);
      k_gemm_tdm<1,__bf16><<<g, 128, 0, stream>>>(ffninb, f1T[l], LW(l,9), midb,
                                                  MROWS, 4*DM, DM, 4*DM);
    }
    gemm_big(midb, f2T[l], LW(l,11), ffnf, MROWS, DM, 4*DM, DM);
    k_addln<<<MROWS, 256, 0, stream>>>(prf, ffnf, LW(l,30), LW(l,31), prf, prb);
    // ---- sequential scan (persistent single block, WMMA inner GEMMs)
    k_scan<<<1, 256, 0, stream>>>(prf,
        taWqT[l], taWkT[l], taWvT[l], taWoT[l], p2sT[l], tr1T[l], tr2T[l],
        LW(l,15), LW(l,16), LW(l,17), LW(l,19), LW(l,25), LW(l,21), LW(l,23),
        LW(l,32), LW(l,33), LW(l,34), LW(l,35),
        hf_g, hb_g, kbufg, vbufg);
  }

  // ---- final LN + decoder (TDM-staged, writes straight into d_out)
  k_addln<<<MROWS, 256, 0, stream>>>(prf, nullptr, lnf_g, lnf_b, lnff, lnfb);
  {
    dim3 g(VPAD/64, MROWS/64);
    k_gemm_tdm<0,float><<<g, 128, 0, stream>>>(lnfb, decWT, dec_b, (float*)d_out,
                                               MROWS, VOCAB, DM, VPAD);
  }
  k_copy_h<<<4, 256, 0, stream>>>(hf_g, (float*)d_out + (size_t)MROWS*VOCAB);
}
